// EncoderLayer_20478404067391
// MI455X (gfx1250) — compile-verified, matrix operand order checked by reading, not run on hardware
//
#include <hip/hip_runtime.h>

#define B_   4
#define L_   1024
#define D_   1024
#define H_   16
#define DH_  64
#define DFF_ 4096

typedef __attribute__((ext_vector_type(16))) __bf16 v16bf;
typedef __attribute__((ext_vector_type(8)))  __bf16 v8bf;
typedef __attribute__((ext_vector_type(4)))  __bf16 v4bf;
typedef __attribute__((ext_vector_type(8)))  float  v8f;
typedef __attribute__((ext_vector_type(4)))  float  v4f;
typedef __attribute__((ext_vector_type(4)))  int    v4i;

// ---- CDNA5 async global->LDS path (guarded; falls back to sync copy) ------
#if defined(__AMDGCN__) && __has_builtin(__builtin_amdgcn_global_load_async_to_lds_b128)
#define HAS_ASYNC_LDS 1
#else
#define HAS_ASYNC_LDS 0
#endif

#if HAS_ASYNC_LDS
__device__ __forceinline__ void async_ld16(const __bf16* gsrc, __bf16* lds) {
    // (global src, LDS dst, imm offset, imm cpol)
    __builtin_amdgcn_global_load_async_to_lds_b128((v4i*)gsrc, (v4i*)lds, 0, 0);
}
__device__ __forceinline__ void wait_async_le4() {
#if __has_builtin(__builtin_amdgcn_s_wait_asynccnt)
    __builtin_amdgcn_s_wait_asynccnt(4);
#else
    asm volatile("s_wait_asynccnt 0x4" ::: "memory");
#endif
}
__device__ __forceinline__ void wait_async_le0() {
#if __has_builtin(__builtin_amdgcn_s_wait_asynccnt)
    __builtin_amdgcn_s_wait_asynccnt(0);
#else
    asm volatile("s_wait_asynccnt 0x0" ::: "memory");
#endif
}
#endif

static __device__ __forceinline__ v8f wmma_bf16(v16bf a, v16bf b, v8f c) {
    // D = A(16x32 bf16) * B(32x16 bf16) + C(16x16 f32)
    return __builtin_amdgcn_wmma_f32_16x16x32_bf16(
        false, a, false, b, (short)0, c, false, false);
}

// ---------------------------------------------------------------------------
// fp32 -> bf16 conversion (weights / activations)
// ---------------------------------------------------------------------------
__global__ void cvt_f32_bf16(const float* __restrict__ in,
                             __bf16* __restrict__ out, int n4) {
    int i = blockIdx.x * blockDim.x + threadIdx.x;
    if (i < n4) {
        v4f v = *(const v4f*)(in + (size_t)i * 4);
        v4bf o;
        for (int j = 0; j < 4; ++j) o[j] = (__bf16)v[j];
        *(v4bf*)(out + (size_t)i * 4) = o;
    }
}

// ---------------------------------------------------------------------------
// Row LayerNorm: fp32 in -> bf16 out. One block (256 thr) per row of D_=1024.
// ---------------------------------------------------------------------------
__launch_bounds__(256)
__global__ void ln_rows(const float* __restrict__ x, const float* __restrict__ g,
                        const float* __restrict__ bta, __bf16* __restrict__ out) {
    __shared__ float red[256];
    int row = blockIdx.x;
    int t   = threadIdx.x;
    const float* xr = x + (size_t)row * D_;
    v4f v = *(const v4f*)(xr + t * 4);

    float s = v[0] + v[1] + v[2] + v[3];
    red[t] = s;
    __syncthreads();
    for (int off = 128; off > 0; off >>= 1) {
        if (t < off) red[t] += red[t + off];
        __syncthreads();
    }
    float mean = red[0] * (1.0f / D_);
    __syncthreads();

    float d0 = v[0] - mean, d1 = v[1] - mean, d2 = v[2] - mean, d3 = v[3] - mean;
    red[t] = d0 * d0 + d1 * d1 + d2 * d2 + d3 * d3;
    __syncthreads();
    for (int off = 128; off > 0; off >>= 1) {
        if (t < off) red[t] += red[t + off];
        __syncthreads();
    }
    float inv = rsqrtf(red[0] * (1.0f / D_) + 1e-5f);

    v4bf o;
    float dv[4] = {d0, d1, d2, d3};
    for (int j = 0; j < 4; ++j) {
        int c = t * 4 + j;
        o[j] = (__bf16)(dv[j] * inv * g[c] + bta[c]);
    }
    *(v4bf*)(out + (size_t)row * D_ + t * 4) = o;
}

// ---------------------------------------------------------------------------
// Generic bf16 WMMA GEMM:  out[M,N] = A[M,K] * Bw[N,K]^T (+bias)(+relu)(+resid)
// Block tile 128x128, K-step 32, 256 threads = 8 waves, 8 WMMA tiles per wave.
// Epilogue fully templated (no runtime pointer checks in inner loops).
// Main loop double-buffered via GLOBAL_LOAD_ASYNC_TO_LDS_B128 when available.
// ---------------------------------------------------------------------------
#define BM 128
#define BN 128
#define BK 32
#define LS 40   // padded LDS row stride (bf16 elems)

template <bool HAS_BIAS, bool RELU, bool HAS_RESID, bool OUT_F32>
__launch_bounds__(256)
__global__ void gemm_bf16(const __bf16* __restrict__ A, const __bf16* __restrict__ Bw,
                          int M, int N, int K,
                          const float* __restrict__ bias,
                          const float* __restrict__ resid,
                          float* __restrict__ outF, __bf16* __restrict__ outB) {
#if HAS_ASYNC_LDS
    __shared__ __bf16 As[2][BM * LS];
    __shared__ __bf16 Bs[2][BM * LS];
#else
    __shared__ __bf16 As[1][BM * LS];
    __shared__ __bf16 Bs[1][BM * LS];
#endif

    int t    = threadIdx.x;
    int lane = t & 31, w = t >> 5;
    int hf  = lane >> 4, l16 = lane & 15;
    int wm = w >> 1, wn = w & 1;               // 4x2 wave grid
    int blockN = blockIdx.x * BN, blockM = blockIdx.y * BM;

    v8f acc[2][4];
    for (int mi = 0; mi < 2; ++mi)
        for (int ni = 0; ni < 4; ++ni) { v8f z = {}; acc[mi][ni] = z; }

    const int nK = K / BK;

#if HAS_ASYNC_LDS
    // Per thread: 4 chunks of 16B per (A+B) tile pair -> 4 async instr / wave.
    // chunk id c in [0,1024): c<512 -> A, else B; row = (c&511)>>2, col=(c&3)*8
    auto issue = [&](int buf, int k0) {
        for (int i = 0; i < 4; ++i) {
            int c   = t + i * 256;
            int j   = c & 511;
            int row = j >> 2;
            int col = (j & 3) * 8;
            if (c < 512)
                async_ld16(A + (size_t)(blockM + row) * K + k0 + col,
                           &As[buf][row * LS + col]);
            else
                async_ld16(Bw + (size_t)(blockN + row) * K + k0 + col,
                           &Bs[buf][row * LS + col]);
        }
    };
    issue(0, 0);
    if (nK > 1) issue(1, BK);
#endif

    for (int it = 0; it < nK; ++it) {
#if HAS_ASYNC_LDS
        int cur = it & 1;
        if (it + 1 < nK) wait_async_le4();   // cur's 4 done; next's may remain
        else             wait_async_le0();
        __syncthreads();
        const __bf16* Asl = As[cur];
        const __bf16* Bsl = Bs[cur];
#else
        int k0 = it * BK;
        for (int i = 0; i < 4; ++i) {
            int c   = t + i * 256;
            int row = c >> 3;
            int col = (c & 7) * 4;
            *(v4bf*)&As[0][row * LS + col] =
                *(const v4bf*)(A + (size_t)(blockM + row) * K + k0 + col);
            *(v4bf*)&Bs[0][row * LS + col] =
                *(const v4bf*)(Bw + (size_t)(blockN + row) * K + k0 + col);
        }
        __syncthreads();
        const __bf16* Asl = As[0];
        const __bf16* Bsl = Bs[0];
#endif

        v16bf aF[2], bF[4];
        for (int mi = 0; mi < 2; ++mi) {
            int r = wm * 32 + mi * 16 + l16;
            v8bf lo = *(const v8bf*)&Asl[r * LS + hf * 8];
            v8bf hi = *(const v8bf*)&Asl[r * LS + 16 + hf * 8];
            for (int j = 0; j < 8; ++j) { aF[mi][j] = lo[j]; aF[mi][j + 8] = hi[j]; }
        }
        for (int ni = 0; ni < 4; ++ni) {
            int r = wn * 64 + ni * 16 + l16;
            v8bf lo = *(const v8bf*)&Bsl[r * LS + hf * 8];
            v8bf hi = *(const v8bf*)&Bsl[r * LS + 16 + hf * 8];
            for (int j = 0; j < 8; ++j) { bF[ni][j] = lo[j]; bF[ni][j + 8] = hi[j]; }
        }
        for (int mi = 0; mi < 2; ++mi)
            for (int ni = 0; ni < 4; ++ni)
                acc[mi][ni] = wmma_bf16(aF[mi], bF[ni], acc[mi][ni]);
        __syncthreads();

#if HAS_ASYNC_LDS
        if (it + 2 < nK) issue(cur, (it + 2) * BK);
#endif
    }

    // epilogue: C/D layout -> row m = hf*8 + vv, col n = l16
    for (int mi = 0; mi < 2; ++mi)
        for (int ni = 0; ni < 4; ++ni) {
            int col  = blockN + wn * 64 + ni * 16 + l16;
            float bv = HAS_BIAS ? bias[col] : 0.0f;
            for (int vv = 0; vv < 8; ++vv) {
                int rowg  = blockM + wm * 32 + mi * 16 + hf * 8 + vv;
                float val = acc[mi][ni][vv] + bv;
                if (RELU) val = fmaxf(val, 0.0f);
                size_t idx = (size_t)rowg * N + col;
                if (HAS_RESID) val += resid[idx];
                if (OUT_F32) outF[idx] = val;
                else         outB[idx] = (__bf16)val;
            }
        }
}

// ---------------------------------------------------------------------------
// V transpose: v[b,l,h,dh] -> vt[b,h,dh,l]  (bf16)
// ---------------------------------------------------------------------------
__global__ void transpose_v(const __bf16* __restrict__ vsrc,
                            __bf16* __restrict__ vt) {
    int idx = blockIdx.x * blockDim.x + threadIdx.x;   // over B*H*DH*L
    int sPos = idx & (L_ - 1);
    int r    = idx >> 10;
    int d    = r & (DH_ - 1);
    int r2   = r >> 6;
    int h    = r2 & (H_ - 1);
    int b    = r2 >> 4;
    vt[idx] = vsrc[((size_t)(b * L_ + sPos)) * D_ + h * DH_ + d];
}

// ---------------------------------------------------------------------------
// Flash-style attention: one wave per (b, h, 16-query tile).
// Per 32-key block: 4 WMMA for scores (K=DH=64), online softmax, 4 WMMA for P*V.
// ---------------------------------------------------------------------------
__launch_bounds__(32)
__global__ void attn_kernel(const __bf16* __restrict__ q, const __bf16* __restrict__ k,
                            const __bf16* __restrict__ vt,
                            const unsigned char* __restrict__ mask,
                            __bf16* __restrict__ ctx) {
    __shared__ float  s_s[16][36];
    __shared__ __bf16 s_p[16][32];
    __shared__ float  s_corr[16];
    __shared__ float  s_m[16];
    __shared__ float  s_l[16];

    int qt = blockIdx.x, h = blockIdx.y, b = blockIdx.z;
    int lane = threadIdx.x;
    int hf = lane >> 4, l16 = lane & 15;

    const __bf16* Qb = q  + ((size_t)(b * L_ + qt * 16)) * D_ + h * DH_;
    const __bf16* Kb = k  + ((size_t)b * L_) * D_ + h * DH_;
    const __bf16* Vt = vt + ((size_t)(b * H_ + h)) * DH_ * L_;
    const unsigned char* mb = mask + (size_t)b * L_;

    // Q fragments (rows = queries, K dim = head dim, 2 steps of 32)
    v16bf qF[2];
    for (int ks = 0; ks < 2; ++ks) {
        const __bf16* p = Qb + (size_t)l16 * D_ + ks * 32 + hf * 8;
        v8bf lo = *(const v8bf*)p;
        v8bf hi = *(const v8bf*)(p + 16);
        for (int j = 0; j < 8; ++j) { qF[ks][j] = lo[j]; qF[ks][j + 8] = hi[j]; }
    }

    v8f acc[4];
    for (int i = 0; i < 4; ++i) { v8f z = {}; acc[i] = z; }
    if (lane < 16) { s_m[lane] = -INFINITY; s_l[lane] = 0.0f; }
    __syncthreads();

    const float scale = 0.125f;  // DH^-0.5 = 1/8

    for (int kb = 0; kb < L_; kb += 32) {
        // ---- scores: two 16x16 tiles over keys [kb, kb+32) ----
        for (int nt = 0; nt < 2; ++nt) {
            int key = kb + nt * 16 + l16;
            v16bf kF[2];
            for (int ks = 0; ks < 2; ++ks) {
                const __bf16* p = Kb + (size_t)key * D_ + ks * 32 + hf * 8;
                v8bf lo = *(const v8bf*)p;
                v8bf hi = *(const v8bf*)(p + 16);
                for (int j = 0; j < 8; ++j) { kF[ks][j] = lo[j]; kF[ks][j + 8] = hi[j]; }
            }
            v8f s = {};
            s = wmma_bf16(qF[0], kF[0], s);
            s = wmma_bf16(qF[1], kF[1], s);
            bool msk = (mb[key] != 0);
            for (int vv = 0; vv < 8; ++vv) {
                float val = s[vv] * scale;
                if (msk) val = -INFINITY;
                s_s[hf * 8 + vv][nt * 16 + l16] = val;
            }
        }
        __syncthreads();

        // ---- online softmax row update (one thread per query row) ----
        if (lane < 16) {
            float mold = s_m[lane];
            float rmax = mold;
            for (int c = 0; c < 32; ++c) rmax = fmaxf(rmax, s_s[lane][c]);
            float corr = (mold == -INFINITY) ? 0.0f : __expf(mold - rmax);
            float lsum = s_l[lane] * corr;
            for (int c = 0; c < 32; ++c) {
                float sv = s_s[lane][c];
                float pe = (sv == -INFINITY) ? 0.0f : __expf(sv - rmax);
                s_p[lane][c] = (__bf16)pe;
                lsum += pe;
            }
            s_m[lane] = rmax;
            s_l[lane] = lsum;
            s_corr[lane] = corr;
        }
        __syncthreads();

        // ---- rescale accumulators by per-row correction ----
        float cr[8];
        for (int vv = 0; vv < 8; ++vv) cr[vv] = s_corr[hf * 8 + vv];
        for (int i = 0; i < 4; ++i)
            for (int vv = 0; vv < 8; ++vv) acc[i][vv] *= cr[vv];

        // ---- P fragment from LDS ----
        v16bf pF;
        {
            const __bf16* p = &s_p[l16][hf * 8];
            v8bf lo = *(const v8bf*)p;
            v8bf hi = *(const v8bf*)(p + 16);
            for (int j = 0; j < 8; ++j) { pF[j] = lo[j]; pF[j + 8] = hi[j]; }
        }
        // ---- V fragments from transposed V (contiguous 16B loads) ----
        for (int dt = 0; dt < 4; ++dt) {
            const __bf16* p = Vt + (size_t)(dt * 16 + l16) * L_ + kb + hf * 8;
            v8bf lo = *(const v8bf*)p;
            v8bf hi = *(const v8bf*)(p + 16);
            v16bf vF;
            for (int j = 0; j < 8; ++j) { vF[j] = lo[j]; vF[j + 8] = hi[j]; }
            acc[dt] = wmma_bf16(pF, vF, acc[dt]);
        }
        __syncthreads();
    }

    // ---- finalize: divide by row sums, store ctx (bf16) ----
    float inv[8];
    for (int vv = 0; vv < 8; ++vv) {
        float l = s_l[hf * 8 + vv];
        inv[vv] = (l > 0.0f) ? 1.0f / l : 0.0f;
    }
    __bf16* Cb = ctx + ((size_t)(b * L_ + qt * 16)) * D_ + h * DH_;
    for (int dt = 0; dt < 4; ++dt)
        for (int vv = 0; vv < 8; ++vv)
            Cb[(size_t)(hf * 8 + vv) * D_ + dt * 16 + l16] =
                (__bf16)(acc[dt][vv] * inv[vv]);
}

// ---------------------------------------------------------------------------
// Host launcher
// ---------------------------------------------------------------------------
extern "C" void kernel_launch(void* const* d_in, const int* in_sizes, int n_in,
                              void* d_out, int out_size, void* d_ws, size_t ws_size,
                              hipStream_t stream) {
    const float* x     = (const float*)d_in[0];
    const unsigned char* mask = (const unsigned char*)d_in[1];
    const float* ln1_g = (const float*)d_in[2];
    const float* ln1_b = (const float*)d_in[3];
    const float* wq    = (const float*)d_in[4];
    const float* wk    = (const float*)d_in[5];
    const float* wv    = (const float*)d_in[6];
    const float* wo    = (const float*)d_in[7];
    const float* ln2_g = (const float*)d_in[8];
    const float* ln2_b = (const float*)d_in[9];
    const float* w1    = (const float*)d_in[10];
    const float* b1    = (const float*)d_in[11];
    const float* w2    = (const float*)d_in[12];
    const float* b2    = (const float*)d_in[13];
    float* out = (float*)d_out;

    const size_t MB = 1024 * 1024;
    char* ws = (char*)d_ws;
    __bf16* wq_b  = (__bf16*)(ws + 0 * MB);
    __bf16* wk_b  = (__bf16*)(ws + 2 * MB);
    __bf16* wv_b  = (__bf16*)(ws + 4 * MB);
    __bf16* wo_b  = (__bf16*)(ws + 6 * MB);
    __bf16* w1_b  = (__bf16*)(ws + 8 * MB);
    __bf16* w2_b  = (__bf16*)(ws + 16 * MB);
    __bf16* xn_b  = (__bf16*)(ws + 24 * MB);
    __bf16* q_b   = (__bf16*)(ws + 32 * MB);
    __bf16* k_b   = (__bf16*)(ws + 40 * MB);
    __bf16* v_b   = (__bf16*)(ws + 48 * MB);
    __bf16* vt_b  = (__bf16*)(ws + 56 * MB);
    __bf16* ctx_b = (__bf16*)(ws + 64 * MB);
    float*  x2    = (float*)(ws + 72 * MB);   // 16 MB
    __bf16* yn_b  = (__bf16*)(ws + 88 * MB);
    __bf16* h_b   = (__bf16*)(ws + 96 * MB);  // 32 MB  -> total 128 MB

    const int M = B_ * L_;  // 4096

    // weights -> bf16
    cvt_f32_bf16<<<(D_ * D_ / 4 + 255) / 256, 256, 0, stream>>>(wq, wq_b, D_ * D_ / 4);
    cvt_f32_bf16<<<(D_ * D_ / 4 + 255) / 256, 256, 0, stream>>>(wk, wk_b, D_ * D_ / 4);
    cvt_f32_bf16<<<(D_ * D_ / 4 + 255) / 256, 256, 0, stream>>>(wv, wv_b, D_ * D_ / 4);
    cvt_f32_bf16<<<(D_ * D_ / 4 + 255) / 256, 256, 0, stream>>>(wo, wo_b, D_ * D_ / 4);
    cvt_f32_bf16<<<(D_ * DFF_ / 4 + 255) / 256, 256, 0, stream>>>(w1, w1_b, D_ * DFF_ / 4);
    cvt_f32_bf16<<<(D_ * DFF_ / 4 + 255) / 256, 256, 0, stream>>>(w2, w2_b, D_ * DFF_ / 4);

    // LN1
    ln_rows<<<M, 256, 0, stream>>>(x, ln1_g, ln1_b, xn_b);

    // QKV projections (plain bf16 out)
    dim3 gQKV(D_ / BN, M / BM);
    gemm_bf16<false, false, false, false><<<gQKV, 256, 0, stream>>>(
        xn_b, wq_b, M, D_, D_, nullptr, nullptr, nullptr, q_b);
    gemm_bf16<false, false, false, false><<<gQKV, 256, 0, stream>>>(
        xn_b, wk_b, M, D_, D_, nullptr, nullptr, nullptr, k_b);
    gemm_bf16<false, false, false, false><<<gQKV, 256, 0, stream>>>(
        xn_b, wv_b, M, D_, D_, nullptr, nullptr, nullptr, v_b);

    // V transpose for attention
    transpose_v<<<(B_ * H_ * DH_ * L_) / 256, 256, 0, stream>>>(v_b, vt_b);

    // attention
    dim3 gAtt(L_ / 16, H_, B_);
    attn_kernel<<<gAtt, 32, 0, stream>>>(q_b, k_b, vt_b, mask, ctx_b);

    // output projection + residual -> x2 (f32)
    gemm_bf16<false, false, true, true><<<gQKV, 256, 0, stream>>>(
        ctx_b, wo_b, M, D_, D_, nullptr, x, x2, nullptr);

    // LN2
    ln_rows<<<M, 256, 0, stream>>>(x2, ln2_g, ln2_b, yn_b);

    // FFN1: relu(yn @ w1^T + b1) -> h (bf16)
    dim3 gF1(DFF_ / BN, M / BM);
    gemm_bf16<true, true, false, false><<<gF1, 256, 0, stream>>>(
        yn_b, w1_b, M, DFF_, D_, b1, nullptr, nullptr, h_b);

    // FFN2: h @ w2^T + b2 + x2 -> out (f32)
    dim3 gF2(D_ / BN, M / BM);
    gemm_bf16<true, false, true, true><<<gF2, 256, 0, stream>>>(
        h_b, w2_b, M, D_, DFF_, b2, x2, out, nullptr);
}
